// LeakyRU_2645699854835
// MI455X (gfx1250) — compile-verified, matrix operand order checked by reading, not run on hardware
//
#include <hip/hip_runtime.h>
#include <math.h>

// LeakyRU: T=2048, B=16, I=1024, H=1024
#define T_STEPS 2048
#define BATCH   16
#define IDIM    1024
#define HDIM    1024
#define NWG     16      // workgroups cooperating in the scan (one 64-col slice each)

typedef __attribute__((ext_vector_type(8)))  float          v8f;
typedef __attribute__((ext_vector_type(16))) __bf16         v16bf;
typedef __attribute__((ext_vector_type(8)))  unsigned int   v8u;
typedef __attribute__((ext_vector_type(2)))  __bf16         v2bf;
typedef int v4i __attribute__((vector_size(4 * sizeof(int))));

#if defined(__has_builtin)
#if __has_builtin(__builtin_amdgcn_cvt_pk_bf16_f32)
#define HAVE_CVT_PK_BF16 1
#endif
#if __has_builtin(__builtin_amdgcn_global_load_async_to_lds_b128)
#define HAVE_ASYNC_LDS 1
#endif
#endif

// ---- helpers -------------------------------------------------------------

__device__ __forceinline__ unsigned short f2bf(float f) {
  __bf16 h = (__bf16)f;     // let the backend pick the hw cvt lowering
  return __builtin_bit_cast(unsigned short, h);
}
// two fp32 -> packed bf16 pair in one dword
__device__ __forceinline__ unsigned pack2(float a, float b) {
#ifdef HAVE_CVT_PK_BF16
  v2bf p = __builtin_amdgcn_cvt_pk_bf16_f32(a, b);
  return __builtin_bit_cast(unsigned, p);
#else
  v2bf p;
  p[0] = (__bf16)a;
  p[1] = (__bf16)b;
  return __builtin_bit_cast(unsigned, p);
#endif
}

#ifdef HAVE_ASYNC_LDS
// builtin wants addrspace(1) / addrspace(3) pointers to v4i.
typedef __attribute__((address_space(1))) v4i* g_v4i_p;
typedef __attribute__((address_space(3))) v4i* l_v4i_p;
__device__ __forceinline__ g_v4i_p to_g(const void* p) {
  return (g_v4i_p)(unsigned long long)p;               // global: same 64-bit value
}
__device__ __forceinline__ l_v4i_p to_l(void* p) {
  return (l_v4i_p)(unsigned int)(unsigned long long)p; // LDS: low 32 bits = offset
}
#endif

__device__ __forceinline__ void wait_asynccnt0() {
#if defined(__has_builtin)
#if __has_builtin(__builtin_amdgcn_s_wait_asynccnt)
  __builtin_amdgcn_s_wait_asynccnt(0);
  return;
#else
  asm volatile("s_wait_asynccnt 0" ::: "memory");
  return;
#endif
#else
  asm volatile("s_wait_asynccnt 0" ::: "memory");
#endif
}

// Device-scope sense barrier across NWG workgroups (persistent cooperative scan).
__device__ __forceinline__ void grid_barrier(unsigned int* bar, unsigned int round) {
  __threadfence();
  __syncthreads();
  if (threadIdx.x == 0) {
    unsigned prev = __hip_atomic_fetch_add(&bar[0], 1u, __ATOMIC_ACQ_REL,
                                           __HIP_MEMORY_SCOPE_AGENT);
    if (prev + 1u == (round + 1u) * NWG) {
      __hip_atomic_store(&bar[1], round + 1u, __ATOMIC_RELEASE,
                         __HIP_MEMORY_SCOPE_AGENT);
    }
    while (__hip_atomic_load(&bar[1], __ATOMIC_ACQUIRE,
                             __HIP_MEMORY_SCOPE_AGENT) < round + 1u) {
      __builtin_amdgcn_s_sleep(2);
    }
  }
  __syncthreads();
  __threadfence();
}

// ---- kernel 0: pack W_in (rows 2H..3H of weight_ih) into bf16 B-operand layout
// B-operand per (ntile, ktile): 32 lanes x 16 bf16; lane -> {N = lane%16, K-half = lane/16}
__global__ void pack_win_kernel(const float* __restrict__ wih,
                                unsigned int* __restrict__ pw) {
  int idx  = blockIdx.x * blockDim.x + threadIdx.x;   // 64*32*32 = 65536 entries
  int lane = idx & 31;
  int kt   = (idx >> 5) & 31;
  int nt   = idx >> 10;                               // 0..63
  int n    = nt * 16 + (lane & 15);
  int hi   = lane >> 4;
  const float* row = wih + (size_t)(2 * HDIM + n) * IDIM;
  int k0 = kt * 32 + 8 * hi;
  unsigned int* dst = pw + (size_t)((nt * 32 + kt) * 32 + lane) * 8;  // 8 dwords = 16 bf16
#pragma unroll
  for (int j = 0; j < 4; ++j)
    dst[j] = pack2(row[k0 + 2 * j], row[k0 + 2 * j + 1]);
#pragma unroll
  for (int j = 0; j < 4; ++j)
    dst[4 + j] = pack2(row[k0 + 16 + 2 * j], row[k0 + 16 + 2 * j + 1]);
}

// ---- kernel 1: pre = x @ W_in^T + bias_n  (written into d_out's y region) --
// grid (2048, 4), block 128: 4 waves/block; each wave computes FOUR 16x16
// N-tiles against one shared A tile, amortizing the fp32->bf16 A conversion
// over 4 WMMAs. HBM traffic ~272 MB => ~12 us at 23.3 TB/s (memory-bound).
__global__ void gemm_pre_kernel(const float* __restrict__ x,
                                const unsigned short* __restrict__ pw,
                                const float* __restrict__ bias_n,
                                float* __restrict__ y) {
  int wave = threadIdx.x >> 5;
  int lane = threadIdx.x & 31;
  int li = lane & 15, hi = lane >> 4;
  int mt  = blockIdx.x;                     // 0..2047  (rows of flattened (t,b))
  int nt0 = blockIdx.y * 16 + wave * 4;     // first of 4 column tiles (0..63)
  const float* xrow = x + (size_t)(mt * 16 + li) * IDIM;
  const unsigned short* wb = pw + (size_t)nt0 * 32 * 32 * 16;

  v8f acc0 = {}, acc1 = {}, acc2 = {}, acc3 = {};
  for (int kt = 0; kt < 32; ++kt) {
    int k0 = kt * 32 + 8 * hi;
    const float4 f0 = *(const float4*)(xrow + k0);
    const float4 f1 = *(const float4*)(xrow + k0 + 4);
    const float4 f2 = *(const float4*)(xrow + k0 + 16);
    const float4 f3 = *(const float4*)(xrow + k0 + 20);
    v8u au;
    au[0] = pack2(f0.x, f0.y); au[1] = pack2(f0.z, f0.w);
    au[2] = pack2(f1.x, f1.y); au[3] = pack2(f1.z, f1.w);
    au[4] = pack2(f2.x, f2.y); au[5] = pack2(f2.z, f2.w);
    au[6] = pack2(f3.x, f3.y); au[7] = pack2(f3.z, f3.w);
    v16bf a = __builtin_bit_cast(v16bf, au);
    const unsigned short* wk = wb + (size_t)(kt * 32 + lane) * 16;
    v16bf b0 = *(const v16bf*)(wk);
    v16bf b1 = *(const v16bf*)(wk + 1 * 32 * 32 * 16);
    v16bf b2 = *(const v16bf*)(wk + 2 * 32 * 32 * 16);
    v16bf b3 = *(const v16bf*)(wk + 3 * 32 * 32 * 16);
    acc0 = __builtin_amdgcn_wmma_f32_16x16x32_bf16(false, a, false, b0,
                                                   (short)0, acc0, false, false);
    acc1 = __builtin_amdgcn_wmma_f32_16x16x32_bf16(false, a, false, b1,
                                                   (short)0, acc1, false, false);
    acc2 = __builtin_amdgcn_wmma_f32_16x16x32_bf16(false, a, false, b2,
                                                   (short)0, acc2, false, false);
    acc3 = __builtin_amdgcn_wmma_f32_16x16x32_bf16(false, a, false, b3,
                                                   (short)0, acc3, false, false);
  }

  float* ybase = y + (size_t)(mt * 16 + 8 * hi) * HDIM + li;
#pragma unroll
  for (int j = 0; j < 4; ++j) {
    v8f acc = (j == 0) ? acc0 : (j == 1) ? acc1 : (j == 2) ? acc2 : acc3;
    int nt = nt0 + j;
    float bn = bias_n[nt * 16 + li];
#pragma unroll
    for (int v = 0; v < 8; ++v) {
      ybase[(size_t)v * HDIM + nt * 16] = acc[v] + bn;
    }
  }
}

// ---- kernel 2: persistent cooperative scan --------------------------------
// 16 WGs x 128 threads (4 waves). Each WG owns 64 H-columns; its slice of
// r*W_hn^T (diag zeroed) stays in LDS as packed bf16 B-operands (128 KB —
// needs CDNA5's 320 KB/WGP LDS). h lives in REGISTERS (8 fp32/lane, the same
// lane produces and consumes h[b,col]); the only cross-WG exchange is h in
// packed bf16 A-operand form (32 KB, L2-resident, double-buffered). Each step:
//   async-copy packed h -> LDS (GLOBAL_LOAD_ASYNC_TO_LDS_B128, zero convert)
//   32x v_wmma_f32_16x16x32_bf16 (one 32 B LDS read per operand)
//   fp32 leak update in registers; scatter 8 bf16 + 8 fp32(y) stores
//   one device-scope barrier
__global__ void scan_kernel(const float* __restrict__ h0,
                            const float* __restrict__ whh,
                            const float* __restrict__ bias_z,
                            const float* __restrict__ bias_r,
                            float* __restrict__ y,
                            float* __restrict__ hout,
                            unsigned short* __restrict__ hpk,  // 2 x 16384 u16
                            unsigned int* __restrict__ bar) {
  extern __shared__ unsigned short smem[];
  unsigned short* Wp = smem;           // 4*32*32*16 u16 = 128 KB
  unsigned short* As = smem + 65536;   // 32*32*16   u16 =  32 KB
  const int tid  = threadIdx.x;
  const int wg   = blockIdx.x;
  const int wave = tid >> 5;
  const int lane = tid & 31;
  const int li = lane & 15, hi = lane >> 4;

  // Pack r ⊙ W_hn^T (diagonal zeroed) for this WG's 64 columns into LDS.
  for (int e = tid; e < 4096; e += 128) {          // (tile, ktile, lane)
    int ln = e & 31;
    int kt = (e >> 5) & 31;
    int tl = e >> 10;
    int nl = ln & 15, hh = ln >> 4;
    int n = wg * 64 + tl * 16 + nl;
    float r = 1.0f / (1.0f + __expf(-bias_r[n]));
    const float* wrow = whh + (size_t)(2 * HDIM + n) * HDIM;
    int k0 = kt * 32 + 8 * hh;
    unsigned int* dst = (unsigned int*)(Wp + ((tl * 32 + kt) * 32 + ln) * 16);
#pragma unroll
    for (int j = 0; j < 4; ++j) {
      int ka = k0 + 2 * j, kb = ka + 1;
      float va = (ka == n) ? 0.0f : r * wrow[ka];
      float vb = (kb == n) ? 0.0f : r * wrow[kb];
      dst[j] = pack2(va, vb);
    }
#pragma unroll
    for (int j = 0; j < 4; ++j) {
      int ka = k0 + 16 + 2 * j, kb = ka + 1;
      float va = (ka == n) ? 0.0f : r * wrow[ka];
      float vb = (kb == n) ? 0.0f : r * wrow[kb];
      dst[4 + j] = pack2(va, vb);
    }
  }

  const int   n0  = wg * 64 + wave * 16;
  const int   col = n0 + li;                 // this lane's H column
  const float z   = 1.0f / (1.0f + __expf(-bias_z[col]));
  const unsigned short* wbase = Wp + wave * (32 * 32 * 16);

  // Packed A-layout address for h[b, k=col]:  idx = (kt*32 + b + 16*hh)*16 + e
  // with kt=col/32, rr=col%32, hh=(rr>>3)&1, e=(rr&7)+(rr>=16 ? 8 : 0).
  const int rr = col & 31;
  const unsigned abase = (unsigned)(((col >> 5) * 32 + 16 * ((rr >> 3) & 1)) * 16
                                    + (rr & 7) + ((rr >= 16) ? 8 : 0));

  // h0 -> registers + packed bf16 buffer 0 (this lane owns b = 8*hi..8*hi+7).
  float hreg[8];
#pragma unroll
  for (int v = 0; v < 8; ++v) {
    int b = v + 8 * hi;
    hreg[v] = h0[b * HDIM + col];
    hpk[abase + b * 16] = f2bf(hreg[v]);
  }

  grid_barrier(bar, 0);   // packed h0 fully populated device-wide

  for (int t = 0; t < T_STEPS; ++t) {
    const unsigned short* hc = hpk + (t & 1) * 16384;
    unsigned short*       hn = hpk + ((t + 1) & 1) * 16384;

    // Stage packed h_t (32 KB, already bf16 A-layout) into LDS.
#ifdef HAVE_ASYNC_LDS
#pragma unroll
    for (int i = 0; i < 16; ++i) {
      int off = i * 2048 + tid * 16;                 // 128 thr x 16 B per sweep
      __builtin_amdgcn_global_load_async_to_lds_b128(
          to_g((const char*)hc + off), to_l((char*)As + off), 0, 0);
    }
    wait_asynccnt0();
#else
    {
      const uint4* src = (const uint4*)hc;
      uint4*       dst = (uint4*)As;
      for (int i = tid; i < 2048; i += 128) dst[i] = src[i];
    }
#endif
    __syncthreads();

    // h @ (r*W_hn)^T for this wave's 16x16 tile: 32 WMMA over K=1024.
    v8f acc = {};
#pragma unroll 4
    for (int kt = 0; kt < 32; ++kt) {
      v16bf a = *(const v16bf*)(As    + (kt * 32 + lane) * 16);
      v16bf b = *(const v16bf*)(wbase + (kt * 32 + lane) * 16);
      acc = __builtin_amdgcn_wmma_f32_16x16x32_bf16(false, a, false, b,
                                                    (short)0, acc, false, false);
    }

    // n = tanh(pre + acc); h' = z*h + (1-z)*n  (fp32 leak path in registers)
    size_t ybase = (size_t)t * BATCH * HDIM;
#pragma unroll
    for (int v = 0; v < 8; ++v) {
      int b = v + 8 * hi;
      size_t oi = ybase + (size_t)b * HDIM + col;
      float pre  = y[oi];                      // pre was written here by kernel 1
      float nval = tanhf(pre + acc[v]);
      float hnew = z * hreg[v] + (1.0f - z) * nval;
      hreg[v] = hnew;
      y[oi] = hnew;                            // y[t] output (overwrites pre)
      hn[abase + b * 16] = f2bf(hnew);         // bf16 A-layout for next step
    }
    grid_barrier(bar, 1u + (unsigned)t);
  }

  // h_n output straight from registers.
#pragma unroll
  for (int v = 0; v < 8; ++v) {
    int b = v + 8 * hi;
    hout[b * HDIM + col] = hreg[v];
  }
}

// ---- launcher --------------------------------------------------------------

extern "C" void kernel_launch(void* const* d_in, const int* in_sizes, int n_in,
                              void* d_out, int out_size, void* d_ws, size_t ws_size,
                              hipStream_t stream) {
  const float* x      = (const float*)d_in[0];
  const float* h0     = (const float*)d_in[1];
  const float* wih    = (const float*)d_in[2];
  const float* whh    = (const float*)d_in[3];
  const float* bias_z = (const float*)d_in[4];
  const float* bias_n = (const float*)d_in[5];
  const float* bias_r = (const float*)d_in[6];

  float* y    = (float*)d_out;                          // T*B*H
  float* hout = y + (size_t)T_STEPS * BATCH * HDIM;     // B*H

  // workspace: [0,2MB) packed bf16 W_in | [2MB,2MB+64KB) packed bf16 h x2 | barrier
  char*           ws  = (char*)d_ws;
  unsigned short* pw  = (unsigned short*)ws;
  unsigned short* hpk = (unsigned short*)(ws + (2u << 20));
  unsigned int*   bar = (unsigned int*)(ws + (2u << 20) + (64u << 10));

  (void)hipMemsetAsync(bar, 0, 2 * sizeof(unsigned int), stream);

  pack_win_kernel<<<256, 256, 0, stream>>>(wih, (unsigned int*)pw);

  gemm_pre_kernel<<<dim3((T_STEPS * BATCH) / 16, 4), 128, 0, stream>>>(
      x, pw, bias_n, y);

  size_t lds_bytes = (size_t)(65536 + 16384) * sizeof(unsigned short);  // 160 KB
  scan_kernel<<<NWG, 128, lds_bytes, stream>>>(h0, whh, bias_z, bias_r,
                                               y, hout, hpk, bar);
}
